// DiffeqSolver_25305947308456
// MI455X (gfx1250) — matrix operand order repaired and checked by reading, not compiled, not run on hardware
//
#include <hip/hip_runtime.h>
#include <hip/hip_bf16.h>
#include <math.h>
#include <stdint.h>

typedef __bf16 bf16_t;
typedef __attribute__((ext_vector_type(16))) __bf16 v16bf;
typedef __attribute__((ext_vector_type(8)))  float  v8f;

#define B_ 16
#define C_ 128
#define H_ 32
#define W_ 32
#define T_ 25
#define NELEM (B_ * C_ * H_ * W_)        // 2,097,152
#define WFRAG_ELEMS (9 * 4 * 8 * 32 * 16) // 147,456 bf16 per weight tensor

#if defined(__has_builtin)
# if __has_builtin(__builtin_amdgcn_global_load_async_to_lds_b128)
#  define HAVE_ASYNC_LDS 1
# endif
# if __has_builtin(__builtin_amdgcn_s_wait_asynccnt)
#  define HAVE_WAIT_ASYNC 1
# endif
#endif

__device__ __forceinline__ bf16_t f2bf(float f) { return (bf16_t)f; }

#if HAVE_ASYNC_LDS
// Builtin takes pointers to 16-byte int vectors: src AS(1), dst AS(3).
// Generic->AS casts via integer round trip (low 32 bits of a generic LDS
// pointer are the LDS byte offset per CDNA5 aperture rules).
typedef int v4i_t __attribute__((vector_size(16)));
typedef __attribute__((address_space(1))) v4i_t gv4i_t;
typedef __attribute__((address_space(3))) v4i_t lv4i_t;

__device__ __forceinline__ void async_cp16(const void* gsrc, void* ldst) {
  gv4i_t* g = (gv4i_t*)(uintptr_t)gsrc;
  lv4i_t* l = (lv4i_t*)(uint32_t)(uintptr_t)ldst;
  __builtin_amdgcn_global_load_async_to_lds_b128(g, l, 0, 0);
}
#endif

// ---------------------------------------------------------------------------
// Weight pre-swizzle: W[co][ci][3][3] f32 -> bf16 A-fragment layout
// [kh][kw][cc(4)][mt(8)][lane(32)][elem(16)]   (16x32 bf16 A, k = 16*half + elem)
// ---------------------------------------------------------------------------
__global__ __launch_bounds__(256) void prep_weights(const float* __restrict__ w,
                                                    bf16_t* __restrict__ out) {
  int e = blockIdx.x * 256 + threadIdx.x;
  if (e >= WFRAG_ELEMS) return;
  int i    = e & 15;
  int lane = (e >> 4) & 31;
  int mt   = (e >> 9) & 7;
  int cc   = (e >> 12) & 3;
  int kidx = e >> 14;           // 0..8
  int kh = kidx / 3, kw = kidx % 3;
  int half = lane >> 4, m = lane & 15;
  int p = i >> 1, j = i & 1;
  int k  = half * 16 + p * 2 + j;
  int co = mt * 16 + m;
  int ci = cc * 32 + k;
  out[e] = f2bf(w[((co * C_ + ci) * 3 + kh) * 3 + kw]);
}

// ---------------------------------------------------------------------------
// Implicit-GEMM 3x3 SAME conv via v_wmma_f32_16x16x32_bf16.
// Input NHWC bf16. One block = two output rows (b, h0..h0+1):
// 128 co x 64 pixels. 8 waves; wave = co tile; 4 accumulators per wave
// (2 rows x 2 pixel-tiles) so each A fragment feeds 4 WMMAs.
// ---------------------------------------------------------------------------
template <bool FUSE_TANH>
__global__ __launch_bounds__(256) void conv3x3_wmma(
    const bf16_t* __restrict__ xin,    // NHWC bf16
    const bf16_t* __restrict__ wfrag,  // pre-swizzled weights
    const float*  __restrict__ bias,
    float*  __restrict__ out_f32,      // NHWC f32 (if !FUSE_TANH)
    bf16_t* __restrict__ out_bf)       // NHWC bf16 (if FUSE_TANH)
{
  constexpr int PIXB = C_ * 2;        // 256 B per pixel
  constexpr int ROWP = 34 * PIXB;     // 8704 B per padded row
  __shared__ alignas(16) unsigned char smem[4 * ROWP];   // rows h0-1 .. h0+2

  const int tid = threadIdx.x;
  const int b   = blockIdx.x >> 4;
  const int h0  = (blockIdx.x & 15) * 2;
  const uint4 zero4 = make_uint4(0u, 0u, 0u, 0u);

  // ---- stage 4 padded input rows into LDS (pure byte copy: async-capable)
  for (int c = tid; c < 2048; c += 256) {   // 2048 x 16B real chunks
    const int r   = c >> 9;                 // row 0..3 -> hh = h0-1..h0+2
    const int rem = c & 511;
    const int pix = rem >> 4;               // 0..31
    const int sub = rem & 15;
    const int hh  = h0 + r - 1;
    unsigned char* dst = smem + r * ROWP + (pix + 1) * PIXB + sub * 16;
    if (hh >= 0 && hh < H_) {
      const unsigned char* src = (const unsigned char*)xin +
          ((size_t)((b * H_ + hh) * W_ + pix) * C_) * 2 + sub * 16;
#if HAVE_ASYNC_LDS
      async_cp16(src, dst);
#else
      *(uint4*)dst = *(const uint4*)src;
#endif
    } else {
      *(uint4*)dst = zero4;                 // zero row (top/bottom halo)
    }
  }
  if (tid < 128) {                          // left/right zero-pad columns
    const int slot = tid >> 4, sub = tid & 15;
    const int r = slot >> 1;
    const int pix = (slot & 1) ? 33 : 0;
    *(uint4*)(smem + r * ROWP + pix * PIXB + sub * 16) = zero4;
  }
#if HAVE_ASYNC_LDS
# if HAVE_WAIT_ASYNC
  __builtin_amdgcn_s_wait_asynccnt(0);
# else
  asm volatile("s_wait_asynccnt 0" ::: "memory");
# endif
#endif
  __syncthreads();

  const int wave = tid >> 5;                // m-tile: co base = wave*16
  const int lane = tid & 31;
  const int lm   = lane & 15;
  const int lh   = lane >> 4;

  union Frag { v16bf v; uint4 q[2]; };
  v8f accs[4] = { {0.f,0.f,0.f,0.f,0.f,0.f,0.f,0.f},
                  {0.f,0.f,0.f,0.f,0.f,0.f,0.f,0.f},
                  {0.f,0.f,0.f,0.f,0.f,0.f,0.f,0.f},
                  {0.f,0.f,0.f,0.f,0.f,0.f,0.f,0.f} };

  for (int kh = 0; kh < 3; ++kh) {
#pragma unroll
    for (int kw = 0; kw < 3; ++kw) {
#pragma unroll
      for (int cc = 0; cc < 4; ++cc) {
        // A fragment: 32 contiguous bytes per lane from swizzled weights
        const unsigned char* ap = (const unsigned char*)wfrag +
            ((size_t)((((kh * 3 + kw) * 4 + cc) * 8 + wave) * 512) + lane * 16) * 2;
        Frag A;
        A.q[0] = *(const uint4*)ap;
        A.q[1] = *(const uint4*)(ap + 16);
        const int cib = (cc * 32 + lh * 16) * 2;
#pragma unroll
        for (int r = 0; r < 2; ++r) {       // output row r -> input row r+kh
          const unsigned char* rowp = smem + (r + kh) * ROWP;
          const unsigned char* bp0 = rowp + (lm + kw) * PIXB + cib;      // pix 0..15
          const unsigned char* bp1 = rowp + (16 + lm + kw) * PIXB + cib; // pix 16..31
          Frag Bf0, Bf1;
          Bf0.q[0] = *(const uint4*)bp0;  Bf0.q[1] = *(const uint4*)(bp0 + 16);
          Bf1.q[0] = *(const uint4*)bp1;  Bf1.q[1] = *(const uint4*)(bp1 + 16);
          accs[r * 2 + 0] = __builtin_amdgcn_wmma_f32_16x16x32_bf16(
              false, A.v, false, Bf0.v, (short)0, accs[r * 2 + 0], false, false);
          accs[r * 2 + 1] = __builtin_amdgcn_wmma_f32_16x16x32_bf16(
              false, A.v, false, Bf1.v, (short)0, accs[r * 2 + 1], false, false);
        }
      }
    }
  }

  // ---- epilogue: C/D layout -> lane owns (co = wave*16 + lh*8 + e, n = lm)
  const int mbase = wave * 16 + lh * 8;
  float bv[8];
#pragma unroll
  for (int e = 0; e < 8; ++e) bv[e] = bias[mbase + e];

#pragma unroll
  for (int r = 0; r < 2; ++r) {
#pragma unroll
    for (int t = 0; t < 2; ++t) {
      const int pixel = t * 16 + lm;
      const size_t obase =
          (size_t)((b * H_ + (h0 + r)) * W_ + pixel) * C_ + mbase;
#pragma unroll
      for (int e = 0; e < 8; ++e) {
        float v = accs[r * 2 + t][e] + bv[e];
        if constexpr (FUSE_TANH) out_bf[obase + e] = f2bf(tanhf(v));
        else                     out_f32[obase + e] = v;
      }
    }
  }
}

// ---------------------------------------------------------------------------
// Elementwise kernels (NHWC internal state; transpose on emit)
// ---------------------------------------------------------------------------
__global__ __launch_bounds__(256) void init_state(const float* __restrict__ fp,  // NCHW
                                                  float* __restrict__ y,         // NHWC
                                                  bf16_t* __restrict__ ybf,
                                                  float* __restrict__ out) {
  int i = blockIdx.x * 256 + threadIdx.x;
  if (i >= NELEM) return;
  int c = i & 127, w = (i >> 7) & 31, h = (i >> 12) & 31, b = i >> 17;
  float v = fp[((size_t)(b * C_ + c) * H_ + h) * W_ + w];
  y[i] = v;
  ybf[i] = f2bf(v);
  out[((size_t)(b * T_ + 0) * C_ + c) * (H_ * W_) + h * W_ + w] = v;
}

__global__ __launch_bounds__(256) void fuse_stage(const float* __restrict__ y,
                                                  const float* __restrict__ k,
                                                  float* __restrict__ acc,
                                                  bf16_t* __restrict__ stage,
                                                  const float* __restrict__ ts, int s,
                                                  float wj, float acoef, int first) {
  int i = blockIdx.x * 256 + threadIdx.x;
  if (i >= NELEM) return;
  float dt = ts[s + 1] - ts[s];
  float kv = k[i];
  acc[i] = first ? (wj * kv) : (acc[i] + wj * kv);
  stage[i] = f2bf(y[i] + acoef * dt * kv);
}

__global__ __launch_bounds__(256) void finalize_step(float* __restrict__ y,
                                                     bf16_t* __restrict__ ybf,
                                                     const float* __restrict__ k4,
                                                     const float* __restrict__ acc,
                                                     float* __restrict__ out,
                                                     const float* __restrict__ ts, int s) {
  int i = blockIdx.x * 256 + threadIdx.x;
  if (i >= NELEM) return;
  float dt = ts[s + 1] - ts[s];
  float v = y[i] + (dt / 6.0f) * (acc[i] + k4[i]);
  y[i] = v;
  ybf[i] = f2bf(v);
  int c = i & 127, w = (i >> 7) & 31, h = (i >> 12) & 31, b = i >> 17;
  out[((size_t)(b * T_ + (s + 1)) * C_ + c) * (H_ * W_) + h * W_ + w] = v;
}

// ---------------------------------------------------------------------------
extern "C" void kernel_launch(void* const* d_in, const int* in_sizes, int n_in,
                              void* d_out, int out_size, void* d_ws, size_t ws_size,
                              hipStream_t stream) {
  const float* fp = (const float*)d_in[0];
  const float* ts = (const float*)d_in[1];
  const float* W1 = (const float*)d_in[2];
  const float* b1 = (const float*)d_in[3];
  const float* W2 = (const float*)d_in[4];
  const float* b2 = (const float*)d_in[5];
  float* out = (float*)d_out;

  char* ws = (char*)d_ws;
  size_t o = 0;
  auto take = [&](size_t bytes) -> void* {
    void* p = ws + o;
    o += (bytes + 255) & ~(size_t)255;
    return p;
  };
  float*  y     = (float*)take((size_t)NELEM * 4);
  float*  acc   = (float*)take((size_t)NELEM * 4);
  float*  kb    = (float*)take((size_t)NELEM * 4);
  bf16_t* ybf   = (bf16_t*)take((size_t)NELEM * 2);
  bf16_t* stage = (bf16_t*)take((size_t)NELEM * 2);
  bf16_t* mid   = (bf16_t*)take((size_t)NELEM * 2);
  bf16_t* w1r   = (bf16_t*)take((size_t)WFRAG_ELEMS * 2);
  bf16_t* w2r   = (bf16_t*)take((size_t)WFRAG_ELEMS * 2);

  const dim3 ew_grid(NELEM / 256), ew_blk(256);
  const dim3 cv_grid(B_ * H_ / 2), cv_blk(256);   // 256 blocks, 2 rows each

  prep_weights<<<(WFRAG_ELEMS + 255) / 256, 256, 0, stream>>>(W1, w1r);
  prep_weights<<<(WFRAG_ELEMS + 255) / 256, 256, 0, stream>>>(W2, w2r);
  init_state<<<ew_grid, ew_blk, 0, stream>>>(fp, y, ybf, out);

  for (int s = 0; s < T_ - 1; ++s) {
    // k1 = f(y)
    conv3x3_wmma<true ><<<cv_grid, cv_blk, 0, stream>>>(ybf, w1r, b1, nullptr, mid);
    conv3x3_wmma<false><<<cv_grid, cv_blk, 0, stream>>>(mid, w2r, b2, kb, nullptr);
    fuse_stage<<<ew_grid, ew_blk, 0, stream>>>(y, kb, acc, stage, ts, s, 1.f, 0.5f, 1);
    // k2 = f(y + dt/2 * k1)
    conv3x3_wmma<true ><<<cv_grid, cv_blk, 0, stream>>>(stage, w1r, b1, nullptr, mid);
    conv3x3_wmma<false><<<cv_grid, cv_blk, 0, stream>>>(mid, w2r, b2, kb, nullptr);
    fuse_stage<<<ew_grid, ew_blk, 0, stream>>>(y, kb, acc, stage, ts, s, 2.f, 0.5f, 0);
    // k3 = f(y + dt/2 * k2)
    conv3x3_wmma<true ><<<cv_grid, cv_blk, 0, stream>>>(stage, w1r, b1, nullptr, mid);
    conv3x3_wmma<false><<<cv_grid, cv_blk, 0, stream>>>(mid, w2r, b2, kb, nullptr);
    fuse_stage<<<ew_grid, ew_blk, 0, stream>>>(y, kb, acc, stage, ts, s, 2.f, 1.0f, 0);
    // k4 = f(y + dt * k3); y += dt/6 * (k1 + 2k2 + 2k3 + k4); emit t=s+1
    conv3x3_wmma<true ><<<cv_grid, cv_blk, 0, stream>>>(stage, w1r, b1, nullptr, mid);
    conv3x3_wmma<false><<<cv_grid, cv_blk, 0, stream>>>(mid, w2r, b2, kb, nullptr);
    finalize_step<<<ew_grid, ew_blk, 0, stream>>>(y, ybf, kb, acc, out, ts, s);
  }
}
